// glimpse_network_71176198029422
// MI455X (gfx1250) — matrix operand order repaired and checked by reading, not compiled, not run on hardware
//
#include <hip/hip_runtime.h>

typedef __attribute__((ext_vector_type(8)))  __bf16 v8bf;
typedef __attribute__((ext_vector_type(16))) __bf16 v16bf;
typedef __attribute__((ext_vector_type(8)))  float  v8f;

__device__ __forceinline__ float lrelu(float v) { return v >= 0.0f ? v : 0.01f * v; }

// hi/lo bf16 split: a ~= hi + lo, |err| ~ 2^-18 relative (fp32-grade in WMMA).
__device__ __forceinline__ void splitbf(float f, __bf16& hi, __bf16& lo) {
    hi = (__bf16)f;                  // RNE convert
    lo = (__bf16)(f - (float)hi);    // exact residual, RNE convert
}

// ---------------------------------------------------------------------------
// WMMA fragment loaders (bf16 arrays, row-major over K).
// A 16x32 layout: lane(half,m): elems 0..7 <- K[kk+8*half .. +7],
//                               elems 8..15 <- K[kk+16+8*half .. +7]
// B 32x16 layout: same as A (lane(half,n): elems 0..7 <- K[kk+8*half..], 8..15 <- K[kk+16+8*half..])
// ---------------------------------------------------------------------------
__device__ __forceinline__ v16bf frag_a(const __bf16* rowkk, int half) {
    v8bf g0 = *(const v8bf*)(rowkk + 8 * half);
    v8bf g1 = *(const v8bf*)(rowkk + 16 + 8 * half);
    return __builtin_shufflevector(g0, g1, 0, 1, 2, 3, 4, 5, 6, 7,
                                           8, 9, 10, 11, 12, 13, 14, 15);
}
__device__ __forceinline__ v16bf frag_b(const __bf16* rowkk, int half) {
    v8bf g0 = *(const v8bf*)(rowkk + 8 * half);
    v8bf g1 = *(const v8bf*)(rowkk + 16 + 8 * half);
    return __builtin_shufflevector(g0, g1, 0, 1, 2, 3, 4, 5, 6, 7,
                                           8, 9, 10, 11, 12, 13, 14, 15);
}

// ---------------------------------------------------------------------------
// Kernel 1: foveation -> phi written directly as bf16 hi/lo pairs.
// One block per (batch, scale); gather + avg-pool, zero outside the image.
// ---------------------------------------------------------------------------
__global__ void k_foveate(const float* __restrict__ x, const float* __restrict__ l,
                          __bf16* __restrict__ phih, __bf16* __restrict__ phil) {
    const int b = blockIdx.x / 3;
    const int k = blockIdx.x % 3;
    const int size = 256 >> k;   // 256, 128, 64
    const int r    = 4 >> k;     // 4, 2, 1
    const float lx = l[b * 2 + 0];
    const float ly = l[b * 2 + 1];
    const int cx = (int)(0.5f * ((lx + 1.0f) * 1024.0f));  // trunc == astype(int32)
    const int cy = (int)(0.5f * ((ly + 1.0f) * 1024.0f));
    const int x0 = cx - size / 2;
    const int y0 = cy - size / 2;
    const float inv = 1.0f / (float)(r * r);
    const float* __restrict__ xb = x + (size_t)b * 1024u * 1024u;
    const size_t obase = (size_t)b * 12288u + (size_t)k * 4096u;

    for (int t = threadIdx.x; t < 4096; t += blockDim.x) {
        const int gy = t >> 6;
        const int gx = t & 63;
        const int yy0 = y0 + gy * r;
        const int xx0 = x0 + gx * r;
        float s = 0.0f;
        for (int i = 0; i < r; ++i) {
            const int yy = yy0 + i;
            if ((unsigned)yy < 1024u) {
                const float* row = xb + (size_t)yy * 1024u;
                for (int j = 0; j < r; ++j) {
                    const int xx = xx0 + j;
                    if ((unsigned)xx < 1024u) s += row[xx];
                }
            }
        }
        __bf16 hi, lo;
        splitbf(s * inv, hi, lo);
        phih[obase + t] = hi;
        phil[obase + t] = lo;
    }
}

// ---------------------------------------------------------------------------
// Kernel 2: elementwise fp32 -> (bf16 hi, bf16 lo). 8 elements per thread.
// ---------------------------------------------------------------------------
__global__ void k_split(const float* __restrict__ src, __bf16* __restrict__ hi,
                        __bf16* __restrict__ lo, int n8) {
    const int i = blockIdx.x * blockDim.x + threadIdx.x;
    if (i >= n8) return;
    const float4* p = (const float4*)src + (size_t)i * 2;
    float4 f0 = p[0], f1 = p[1];
    const float f[8] = {f0.x, f0.y, f0.z, f0.w, f1.x, f1.y, f1.z, f1.w};
    v8bf vh, vl;
#pragma unroll
    for (int j = 0; j < 8; ++j) {
        __bf16 h, q;
        splitbf(f[j], h, q);
        vh[j] = h;
        vl[j] = q;
    }
    ((v8bf*)hi)[i] = vh;
    ((v8bf*)lo)[i] = vl;
}

// Kernel 3: (W3 + W4) -> (bf16 hi, bf16 lo). 8 elements per thread.
__global__ void k_addsplit(const float* __restrict__ a, const float* __restrict__ b,
                           __bf16* __restrict__ hi, __bf16* __restrict__ lo, int n8) {
    const int i = blockIdx.x * blockDim.x + threadIdx.x;
    if (i >= n8) return;
    const float4* pa = (const float4*)a + (size_t)i * 2;
    const float4* pb = (const float4*)b + (size_t)i * 2;
    float4 a0 = pa[0], a1 = pa[1];
    float4 b0 = pb[0], b1 = pb[1];
    const float f[8] = {a0.x + b0.x, a0.y + b0.y, a0.z + b0.z, a0.w + b0.w,
                        a1.x + b1.x, a1.y + b1.y, a1.z + b1.z, a1.w + b1.w};
    v8bf vh, vl;
#pragma unroll
    for (int j = 0; j < 8; ++j) {
        __bf16 h, q;
        splitbf(f[j], h, q);
        vh[j] = h;
        vl[j] = q;
    }
    ((v8bf*)hi)[i] = vh;
    ((v8bf*)lo)[i] = vl;
}

// ---------------------------------------------------------------------------
// Kernel 4: h = leaky_relu(phi @ W1^T).  M=64, N=512, K=12288.
// One wave per 16x16 tile (128 waves). Pure bf16 loads + 3 WMMAs per K-step:
//   c += ah*bh + ah*bl + al*bh   (split accumulation, fp32 accumulator)
// Epilogue stores h as bf16 hi/lo for GEMM2.
// ---------------------------------------------------------------------------
__global__ void k_gemm1(const __bf16* __restrict__ ph, const __bf16* __restrict__ pl,
                        const __bf16* __restrict__ w1h, const __bf16* __restrict__ w1l,
                        __bf16* __restrict__ hh, __bf16* __restrict__ hl) {
    const int lane = threadIdx.x & 31;
    const int wave = threadIdx.x >> 5;
    const int tile = blockIdx.x * 8 + wave;    // 128 tiles (4 x 32)
    const int tm   = tile >> 5;
    const int tn   = tile & 31;
    const int half = lane >> 4;
    const int lrow = lane & 15;

    const __bf16* __restrict__ ah_row = ph  + (size_t)(tm * 16 + lrow) * 12288u;
    const __bf16* __restrict__ al_row = pl  + (size_t)(tm * 16 + lrow) * 12288u;
    const __bf16* __restrict__ bh_row = w1h + (size_t)(tn * 16 + lrow) * 12288u;
    const __bf16* __restrict__ bl_row = w1l + (size_t)(tn * 16 + lrow) * 12288u;

    v8f c = {};
#pragma unroll 2
    for (int kk = 0; kk < 12288; kk += 32) {
        v16bf ah = frag_a(ah_row + kk, half);
        v16bf al = frag_a(al_row + kk, half);
        v16bf bh = frag_b(bh_row + kk, half);
        v16bf bl = frag_b(bl_row + kk, half);
        c = __builtin_amdgcn_wmma_f32_16x16x32_bf16(false, ah, false, bh, (short)0, c, false, false);
        c = __builtin_amdgcn_wmma_f32_16x16x32_bf16(false, ah, false, bl, (short)0, c, false, false);
        c = __builtin_amdgcn_wmma_f32_16x16x32_bf16(false, al, false, bh, (short)0, c, false, false);
    }
#pragma unroll
    for (int r = 0; r < 8; ++r) {
        const int m = tm * 16 + r + 8 * half;   // C/D layout: VGPR r -> M=r (+8 hi lanes)
        const int n = tn * 16 + lrow;
        const float v = lrelu(c[r]);
        __bf16 hi, lo;
        splitbf(v, hi, lo);
        hh[(size_t)m * 512u + n] = hi;
        hl[(size_t)m * 512u + n] = lo;
    }
}

// ---------------------------------------------------------------------------
// Kernel 5: out = leaky_relu(h @ (W3+W4)^T).  M=64, N=1024, K=512.
// ---------------------------------------------------------------------------
__global__ void k_gemm2(const __bf16* __restrict__ hh, const __bf16* __restrict__ hl,
                        const __bf16* __restrict__ wsh, const __bf16* __restrict__ wsl,
                        float* __restrict__ out) {
    const int lane = threadIdx.x & 31;
    const int wave = threadIdx.x >> 5;
    const int tile = blockIdx.x * 8 + wave;    // 256 tiles (4 x 64)
    const int tm   = tile >> 6;
    const int tn   = tile & 63;
    const int half = lane >> 4;
    const int lrow = lane & 15;

    const __bf16* __restrict__ ah_row = hh  + (size_t)(tm * 16 + lrow) * 512u;
    const __bf16* __restrict__ al_row = hl  + (size_t)(tm * 16 + lrow) * 512u;
    const __bf16* __restrict__ bh_row = wsh + (size_t)(tn * 16 + lrow) * 512u;
    const __bf16* __restrict__ bl_row = wsl + (size_t)(tn * 16 + lrow) * 512u;

    v8f c = {};
#pragma unroll 4
    for (int kk = 0; kk < 512; kk += 32) {
        v16bf ah = frag_a(ah_row + kk, half);
        v16bf al = frag_a(al_row + kk, half);
        v16bf bh = frag_b(bh_row + kk, half);
        v16bf bl = frag_b(bl_row + kk, half);
        c = __builtin_amdgcn_wmma_f32_16x16x32_bf16(false, ah, false, bh, (short)0, c, false, false);
        c = __builtin_amdgcn_wmma_f32_16x16x32_bf16(false, ah, false, bl, (short)0, c, false, false);
        c = __builtin_amdgcn_wmma_f32_16x16x32_bf16(false, al, false, bh, (short)0, c, false, false);
    }
#pragma unroll
    for (int r = 0; r < 8; ++r) {
        const int m = tm * 16 + r + 8 * half;
        const int n = tn * 16 + lrow;
        out[(size_t)m * 1024u + n] = lrelu(c[r]);
    }
}

// ---------------------------------------------------------------------------
// Launch. Workspace layout (64B-aligned chunks), ~30.5 MB total:
//   phih/phil : 64x12288 bf16 each
//   w1h/w1l   : 512x12288 bf16 each
//   wsh/wsl   : 1024x512 bf16 each
//   hh/hl     : 64x512 bf16 each
// ---------------------------------------------------------------------------
extern "C" void kernel_launch(void* const* d_in, const int* in_sizes, int n_in,
                              void* d_out, int out_size, void* d_ws, size_t ws_size,
                              hipStream_t stream) {
    const float* x  = (const float*)d_in[0];   // (64,1,1024,1024)
    const float* l  = (const float*)d_in[1];   // (64,2)
    const float* W1 = (const float*)d_in[2];   // (512,12288)
    // d_in[3] = W2 -> dead code in the reference
    const float* W3 = (const float*)d_in[4];   // (1024,512)
    const float* W4 = (const float*)d_in[5];   // (1024,512)

    char* ws = (char*)d_ws;
    const size_t PHI = (size_t)64 * 12288u * 2u;     // 1.5 MiB
    const size_t W1S = (size_t)512 * 12288u * 2u;    // 12 MiB
    const size_t WSS = (size_t)1024 * 512u * 2u;     // 1 MiB
    const size_t HS  = (size_t)64 * 512u * 2u;       // 64 KiB
    __bf16* phih = (__bf16*)(ws);
    __bf16* phil = (__bf16*)(ws + PHI);
    __bf16* w1h  = (__bf16*)(ws + 2 * PHI);
    __bf16* w1l  = (__bf16*)(ws + 2 * PHI + W1S);
    __bf16* wsh  = (__bf16*)(ws + 2 * PHI + 2 * W1S);
    __bf16* wsl  = (__bf16*)(ws + 2 * PHI + 2 * W1S + WSS);
    __bf16* hh   = (__bf16*)(ws + 2 * PHI + 2 * W1S + 2 * WSS);
    __bf16* hl   = (__bf16*)(ws + 2 * PHI + 2 * W1S + 2 * WSS + HS);

    // Independent prep passes (serialized on stream; each is bandwidth-bound).
    k_split   <<<dim3(3072), dim3(256), 0, stream>>>(W1, w1h, w1l, 512 * 12288 / 8);
    k_addsplit<<<dim3(256),  dim3(256), 0, stream>>>(W3, W4, wsh, wsl, 1024 * 512 / 8);
    k_foveate <<<dim3(192),  dim3(256), 0, stream>>>(x, l, phih, phil);
    // GEMMs (pure bf16 loads + WMMA; W1 fragments re-hit 192MB L2).
    k_gemm1   <<<dim3(16),   dim3(256), 0, stream>>>(phih, phil, w1h, w1l, hh, hl);
    k_gemm2   <<<dim3(32),   dim3(256), 0, stream>>>(hh, hl, wsh, wsl, (float*)d_out);
}
